// QuantumHybrid_11708080849172
// MI455X (gfx1250) — matrix operand order repaired and checked by reading, not compiled, not run on hardware
//
#include <hip/hip_runtime.h>
#include <cstddef>

typedef __attribute__((ext_vector_type(16))) _Float16 v16h;
typedef __attribute__((ext_vector_type(8)))  _Float16 v8h;
typedef __attribute__((ext_vector_type(8)))  float    v8f;

#define FEAT      1280
#define KCHUNKS   40          // 1280 / 32
#define ROWS_PB   128         // rows per block (8 waves * 16)

__global__ __launch_bounds__(256) void qhybrid_fused(
    const float* __restrict__ features,   // (B,1280)
    const float* __restrict__ enc_w,      // (1280,4)
    const float* __restrict__ enc_b,      // (4,)
    const float* __restrict__ q_weights,  // (4,4)
    const float* __restrict__ w1,         // (4,64)
    const float* __restrict__ b1,         // (64,)
    const float* __restrict__ w2,         // (64,3)
    const float* __restrict__ b2,         // (3,)
    float* __restrict__ out)              // (B,3)
{
    __shared__ float    sA[2][ROWS_PB][36]; // double-buffered 128x32 f32 tile (+pad)
    __shared__ _Float16 sBh[4][1288];       // enc_w hi-f16, n-major, k-contig + pad
    __shared__ _Float16 sBl[4][1288];       // enc_w lo-residual f16
    __shared__ float    sX[8][16][4];       // per-wave tanh angles
    __shared__ float    sW1[256];           // (4,64) flat
    __shared__ float    sB1[64];
    __shared__ float    sW2[192];           // (64,3) flat
    __shared__ float    sB2[4];
    __shared__ float    sQc[16], sQs[16];   // cos/sin(q_weights/2)

    const int tid  = threadIdx.x;
    const int lane = tid & 31;
    const int wave = tid >> 5;
    const int hi   = lane >> 4;             // lane group (K-offset selector)
    const int m    = lane & 15;             // A row / C column index
    const int R0   = blockIdx.x * ROWS_PB;
    const unsigned long long fbase = (unsigned long long)features;

    // per-thread staging coordinates (4 b128 transfers per thread per chunk)
    int srow[4], sseg[4];
    unsigned sva[4];
    #pragma unroll
    for (int it = 0; it < 4; ++it) {
        int idx = tid + it * 256;           // 0..1023
        srow[it] = idx >> 3;
        sseg[it] = idx & 7;
        sva[it]  = (unsigned)((size_t)(R0 + srow[it]) * (FEAT * 4) + sseg[it] * 16);
    }

    // issue async copy of chunk 0 into buffer 0 ASAP (before constant staging)
    #pragma unroll
    for (int it = 0; it < 4; ++it) {
        unsigned laddr = (unsigned)(unsigned long long)(void*)&sA[0][srow[it]][sseg[it] * 4];
        unsigned voff  = sva[it];
        asm volatile("global_load_async_to_lds_b128 %0, %1, %2"
                     :: "v"(laddr), "v"(voff), "s"(fbase) : "memory");
    }

    // ---- stage small constants into LDS (enc_w pre-split to f16 hi/lo) ----
    for (int i = tid; i < FEAT * 4; i += 256) {
        const int k = i >> 2, n = i & 3;
        float f = enc_w[i];
        _Float16 h = (_Float16)f;
        sBh[n][k] = h;
        sBl[n][k] = (_Float16)(f - (float)h);
    }
    sW1[tid] = w1[tid];                     // exactly 256 elements
    if (tid < 64)  sB1[tid] = b1[tid];
    if (tid < 192) sW2[tid] = w2[tid];
    if (tid < 3)   sB2[tid] = b2[tid];
    if (tid < 16) {
        float a = 0.5f * q_weights[tid];
        sQc[tid] = cosf(a);
        sQs[tid] = sinf(a);
    }
    __syncthreads();

    // ---- encoder GEMM: C(16x16 per wave) += A(16x32) x B(32x16), split fp32 ----
    v8f acc = {};
    const int arow = wave * 16 + m;
    const int koff = 8 * hi;

    for (int kc = 0; kc < KCHUNKS; ++kc) {
        const int cur = kc & 1;

        // prefetch next chunk into the other buffer, then wait only for current
        if (kc < KCHUNKS - 1) {
            #pragma unroll
            for (int it = 0; it < 4; ++it) {
                unsigned laddr = (unsigned)(unsigned long long)
                                 (void*)&sA[cur ^ 1][srow[it]][sseg[it] * 4];
                unsigned voff  = sva[it] + (unsigned)((kc + 1) * 128);
                asm volatile("global_load_async_to_lds_b128 %0, %1, %2"
                             :: "v"(laddr), "v"(voff), "s"(fbase) : "memory");
            }
            // async loads complete in order: <=4 outstanding => current tile done
            asm volatile("s_wait_asynccnt 0x4" ::: "memory");
        } else {
            asm volatile("s_wait_asynccnt 0x0" ::: "memory");
        }
        __syncthreads();   // current tile visible to all waves

        // A fragment: a[0..7]=A[m][8*hi+0..7], a[8..15]=A[m][16+8*hi+0..7]
        float av[16];
        {
            const float* ap = &sA[cur][arow][0];
            float4 t0 = *reinterpret_cast<const float4*>(ap + koff);
            float4 t1 = *reinterpret_cast<const float4*>(ap + koff + 4);
            float4 t2 = *reinterpret_cast<const float4*>(ap + 16 + koff);
            float4 t3 = *reinterpret_cast<const float4*>(ap + 16 + koff + 4);
            av[0]=t0.x;  av[1]=t0.y;  av[2]=t0.z;  av[3]=t0.w;
            av[4]=t1.x;  av[5]=t1.y;  av[6]=t1.z;  av[7]=t1.w;
            av[8]=t2.x;  av[9]=t2.y;  av[10]=t2.z; av[11]=t2.w;
            av[12]=t3.x; av[13]=t3.y; av[14]=t3.z; av[15]=t3.w;
        }
        v16h a_hi, a_lo;
        #pragma unroll
        for (int e = 0; e < 16; ++e) {
            _Float16 h = (_Float16)av[e];
            a_hi[e] = h;
            a_lo[e] = (_Float16)(av[e] - (float)h);
        }

        // B fragment: b[0..7] = enc_w[kc*32 + 8*hi + 0..7][n], b[8..15] = enc_w[kc*32 + 16 + 8*hi + 0..7][n], n = m (zero for n>=4)
        v16h b_hi = {}, b_lo = {};
        if (m < 4) {
            const int kb = kc * 32 + 8 * hi;
            v8h h0 = *reinterpret_cast<const v8h*>(&sBh[m][kb]);
            v8h h1 = *reinterpret_cast<const v8h*>(&sBh[m][kb + 16]);
            v8h l0 = *reinterpret_cast<const v8h*>(&sBl[m][kb]);
            v8h l1 = *reinterpret_cast<const v8h*>(&sBl[m][kb + 16]);
            #pragma unroll
            for (int e = 0; e < 8; ++e) {
                b_hi[e] = h0[e];  b_hi[8 + e] = h1[e];
                b_lo[e] = l0[e];  b_lo[8 + e] = l1[e];
            }
        }

        // fp32-accurate product via hi/lo split (uniform control flow, EXEC all 1s)
        acc = __builtin_amdgcn_wmma_f32_16x16x32_f16(false, a_hi, false, b_hi,
                                                     (short)0, acc, false, false);
        acc = __builtin_amdgcn_wmma_f32_16x16x32_f16(false, a_lo, false, b_hi,
                                                     (short)0, acc, false, false);
        acc = __builtin_amdgcn_wmma_f32_16x16x32_f16(false, a_hi, false, b_lo,
                                                     (short)0, acc, false, false);
        __syncthreads();   // all waves done reading buf[cur] before it is refilled
    }

    // ---- tanh(x) angles to LDS: c[v] = C[v + 8*hi][n=m] ----
    if (m < 4) {
        float eb = enc_b[m];
        #pragma unroll
        for (int v = 0; v < 8; ++v)
            sX[wave][v + 8 * hi][m] = tanhf(acc[v] + eb);
    }
    __syncthreads();

    // ---- per-sample 4-qubit statevector sim + MLP head (lanes 0..15) ----
    if (lane < 16) {
        const int row = R0 + wave * 16 + lane;
        float4 ang = *reinterpret_cast<const float4*>(&sX[wave][lane][0]);
        float th[4] = {ang.x, ang.y, ang.z, ang.w};

        float sr[16], si[16];
        #pragma unroll
        for (int i = 0; i < 16; ++i) { sr[i] = 0.f; si[i] = 0.f; }
        sr[0] = 1.f;

        // RX(theta) pairs: a' = c*a - i*s*b ; b' = -i*s*a + c*b
        #define RX_WIRE(d, c, s)                                        \
            _Pragma("unroll")                                           \
            for (int i = 0; i < 16; ++i) {                              \
                if ((i & (d)) == 0) {                                   \
                    int j = i | (d);                                    \
                    float ar = sr[i], ai = si[i];                       \
                    float br = sr[j], bi = si[j];                       \
                    sr[i] = (c)*ar + (s)*bi;  si[i] = (c)*ai - (s)*br;  \
                    sr[j] = (c)*br + (s)*ai;  si[j] = (c)*bi - (s)*ar;  \
                }                                                       \
            }

        // AngleEmbedding (RX per wire, per-sample angle)
        #pragma unroll
        for (int w = 0; w < 4; ++w) {
            float c = cosf(0.5f * th[w]);
            float s = sinf(0.5f * th[w]);
            const int d = 8 >> w;
            RX_WIRE(d, c, s)
        }
        // BasicEntanglerLayers: RX (fixed weights) + CNOT ring
        #pragma unroll
        for (int l = 0; l < 4; ++l) {
            #pragma unroll
            for (int w = 0; w < 4; ++w) {
                float c = sQc[l * 4 + w];
                float s = sQs[l * 4 + w];
                const int d = 8 >> w;
                RX_WIRE(d, c, s)
            }
            #pragma unroll
            for (int w = 0; w < 4; ++w) {
                const int dc = 8 >> w;
                const int dt = 8 >> ((w + 1) & 3);
                #pragma unroll
                for (int i = 0; i < 16; ++i) {
                    if ((i & dc) != 0 && (i & dt) == 0) {
                        int j = i | dt;
                        float tr = sr[i]; sr[i] = sr[j]; sr[j] = tr;
                        float ti = si[i]; si[i] = si[j]; si[j] = ti;
                    }
                }
            }
        }
        #undef RX_WIRE

        // <Z_w> expectations
        float ev[4];
        #pragma unroll
        for (int w = 0; w < 4; ++w) {
            const int d = 8 >> w;
            float e = 0.f;
            #pragma unroll
            for (int i = 0; i < 16; ++i) {
                float p = sr[i] * sr[i] + si[i] * si[i];
                e += ((i & d) ? -p : p);
            }
            ev[w] = e;
        }

        // head: relu(q@w1 + b1) @ w2 + b2
        float o0 = sB2[0], o1 = sB2[1], o2 = sB2[2];
        #pragma unroll 8
        for (int j = 0; j < 64; ++j) {
            float h = sB1[j];
            h = fmaf(ev[0], sW1[j],        h);
            h = fmaf(ev[1], sW1[64 + j],   h);
            h = fmaf(ev[2], sW1[128 + j],  h);
            h = fmaf(ev[3], sW1[192 + j],  h);
            h = fmaxf(h, 0.f);
            o0 = fmaf(h, sW2[j * 3 + 0], o0);
            o1 = fmaf(h, sW2[j * 3 + 1], o1);
            o2 = fmaf(h, sW2[j * 3 + 2], o2);
        }
        float* op = out + (size_t)row * 3;
        op[0] = o0; op[1] = o1; op[2] = o2;
    }
}

extern "C" void kernel_launch(void* const* d_in, const int* in_sizes, int n_in,
                              void* d_out, int out_size, void* d_ws, size_t ws_size,
                              hipStream_t stream) {
    (void)n_in; (void)d_ws; (void)ws_size; (void)out_size;
    const float* features  = (const float*)d_in[0];
    const float* enc_w     = (const float*)d_in[1];
    const float* enc_b     = (const float*)d_in[2];
    const float* q_weights = (const float*)d_in[3];
    const float* w1        = (const float*)d_in[4];
    const float* b1        = (const float*)d_in[5];
    const float* w2        = (const float*)d_in[6];
    const float* b2        = (const float*)d_in[7];
    float* out = (float*)d_out;

    const int B = in_sizes[0] / FEAT;          // 65536
    const int nblocks = B / ROWS_PB;           // 512
    qhybrid_fused<<<nblocks, 256, 0, stream>>>(features, enc_w, enc_b, q_weights,
                                               w1, b1, w2, b2, out);
}
